// HGATLayer_27212912788252
// MI455X (gfx1250) — matrix operand order repaired and checked by reading, not compiled, not run on hardware
//
#include <hip/hip_runtime.h>
#include <hip/hip_bf16.h>

typedef __bf16 bf16;
typedef __attribute__((ext_vector_type(16))) __bf16 v16bf;
typedef __attribute__((ext_vector_type(8)))  __bf16 v8bf;
typedef __attribute__((ext_vector_type(8)))  float  v8f;

// Problem constants (match reference)
#define B_   32
#define N_   512
#define NH_  4
#define H_   32
#define DIN_ 128
#define BIGF 9.0e15f
#define INV_SQRT_DK 0.17677669529663687f   // 1/sqrt(32)

// ---------------------------------------------------------------- WMMA helper
__device__ __forceinline__ v8f wmma_bf16(v16bf a, v16bf b, v8f c) {
  // (neg_a, A, neg_b, B, c_mod, C, reuse_a, reuse_b)
  return __builtin_amdgcn_wmma_f32_16x16x32_bf16(false, a, false, b,
                                                 (short)0, c, false, false);
}

// Reductions across one 16-lane half of a wave32 (xor masks 1,2,4,8 stay
// inside each half, so rows 0-7 (lanes 0-15) and 8-15 (lanes 16-31) reduce
// independently).
__device__ __forceinline__ float half16_max(float v) {
#pragma unroll
  for (int m = 1; m < 16; m <<= 1) v = fmaxf(v, __shfl_xor(v, m, 32));
  return v;
}
__device__ __forceinline__ float half16_sum(float v) {
#pragma unroll
  for (int m = 1; m < 16; m <<= 1) v += __shfl_xor(v, m, 32);
  return v;
}
__device__ __forceinline__ int half16_or(int v) {
#pragma unroll
  for (int m = 1; m < 16; m <<= 1) v |= __shfl_xor(v, m, 32);
  return v;
}

// Build a 16-bit A-matrix operand (16x32, K=32) in contiguous order:
// lanes 0-15 hold K 0-15 of row M=lane; lanes 16-31 hold K 16-31.
__device__ __forceinline__ v16bf loadA16(const bf16* rowptr, int kbase, int half) {
  v8bf lo = *(const v8bf*)(rowptr + kbase + half * 16);
  v8bf hi = *(const v8bf*)(rowptr + kbase + half * 16 + 8);
  v16bf a;
#pragma unroll
  for (int i = 0; i < 8; i++) { a[i] = lo[i]; a[8 + i] = hi[i]; }
  return a;
}

// ------------------------------------------------- weight transpose (f32->bf16)
// src[k*N + n]  ->  dst[n*K + k]   (B-operand wants contiguous K per N-column)
__global__ void tr_kernel(const float* __restrict__ src, bf16* __restrict__ dst,
                          int K, int N) {
  int i = blockIdx.x * 256 + threadIdx.x;
  if (i < K * N) {
    int k = i / N, n = i % N;
    dst[(size_t)n * K + k] = (bf16)src[i];
  }
}

// --------------------------------------------------------------- QKV kernel
// out = relu(x @ W + b), x f32 [B,N,128], wT bf16 [128out][128in].
// storeT=0: out[b][n][c] (row-major, feeds A operands later)
// storeT=1: out[b][c][n] (transposed V, feeds P@V B operands)
__global__ void __launch_bounds__(128) qkv_kernel(
    const float* __restrict__ x, const bf16* __restrict__ wT,
    const float* __restrict__ bias, bf16* __restrict__ out, int storeT) {
  int lane = threadIdx.x & 31, wv = threadIdx.x >> 5;
  int half = lane >> 4, li = lane & 15;
  int t = blockIdx.x * 4 + wv;
  int b = t >> 5, qbase = (t & 31) * 16;

  const float* xr = x + ((size_t)(b * N_ + qbase + li)) * DIN_;
  v16bf a[4];
#pragma unroll
  for (int ks = 0; ks < 4; ks++) {
    v8f lo = *(const v8f*)(xr + ks * 32 + half * 16);
    v8f hi = *(const v8f*)(xr + ks * 32 + half * 16 + 8);
#pragma unroll
    for (int i = 0; i < 8; i++) { a[ks][i] = (bf16)lo[i]; a[ks][8 + i] = (bf16)hi[i]; }
  }

#pragma unroll
  for (int nt = 0; nt < 8; nt++) {
    v8f acc = {};
    int n = nt * 16 + li;
    const bf16* wp = wT + (size_t)n * DIN_ + half * 16;
#pragma unroll
    for (int ks = 0; ks < 4; ks++) {
      v16bf bw = *(const v16bf*)(wp + ks * 32);
      acc = wmma_bf16(a[ks], bw, acc);
    }
    float bb = bias[n];
#pragma unroll
    for (int r = 0; r < 8; r++) {
      int row = qbase + r + half * 8;
      float v = acc[r] + bb;
      v = v > 0.f ? v : 0.f;
      if (storeT) out[((size_t)(b * DIN_) + n) * N_ + row] = (bf16)v;
      else        out[((size_t)(b * N_) + row) * DIN_ + n] = (bf16)v;
    }
  }
}

// ---------------------------------------------------------- attention kernel
// Flash-style masked softmax attention per (b, head, 16-query tile).
// q,k: bf16 [B,N,128] row-major.  vT: bf16 [B,128,N].  mask: int32 [B,N,N].
__global__ void __launch_bounds__(128) attn_kernel(
    const bf16* __restrict__ q, const bf16* __restrict__ k,
    const bf16* __restrict__ vT, const int* __restrict__ mask,
    bf16* __restrict__ att, int gate) {
  __shared__ __align__(32) bf16 pbuf[4][16][32];

  int lane = threadIdx.x & 31, wv = threadIdx.x >> 5;
  int half = lane >> 4, li = lane & 15;
  int t = blockIdx.x * 4 + wv;
  int b = t >> 7, h = (t >> 5) & 3, qbase = (t & 31) * 16;

  // Q A-operand (16 rows x K=32 head dims), loaded once
  v16bf aq = loadA16(q + ((size_t)(b * N_ + qbase + li)) * DIN_ + h * H_, 0, half);

  v8f o0 = {}, o1 = {};
  float rmax[8], rsum[8];
  int rhas[8];
#pragma unroll
  for (int r = 0; r < 8; r++) { rmax[r] = -3.0e38f; rsum[r] = 0.f; rhas[r] = 0; }

  const int* mbase = mask + (size_t)b * N_ * N_;

  for (int ks = 0; ks < N_; ks += 32) {
    // S = Q @ K^T for 32 keys (two 16x16 tiles).  B operand: lane = key li,
    // contiguous 16 dims per lane-half.
    const bf16* kb = k + ((size_t)(b * N_ + ks + li)) * DIN_ + h * H_ + half * 16;
    v16bf bk0 = *(const v16bf*)kb;
    v16bf bk1 = *(const v16bf*)(kb + 16 * DIN_);
    if (ks + 32 < N_) __builtin_prefetch(kb + 32 * DIN_, 0, 0);

    v8f z = {};
    v8f s0 = wmma_bf16(aq, bk0, z);
    v8f s1 = wmma_bf16(aq, bk1, z);

    float p0[8], p1[8];
#pragma unroll
    for (int r = 0; r < 8; r++) {
      int qrow = qbase + r + half * 8;
      const int* mr = mbase + (size_t)qrow * N_ + ks + li;
      int m0 = mr[0], m1 = mr[16];
      rhas[r] |= m0 | m1;
      float v0 = m0 ? s0[r] * INV_SQRT_DK : -BIGF;
      float v1 = m1 ? s1[r] * INV_SQRT_DK : -BIGF;
      float tmax = half16_max(fmaxf(v0, v1));
      float nmax = fmaxf(rmax[r], tmax);
      float corr = __expf(rmax[r] - nmax);
      float e0 = __expf(v0 - nmax), e1 = __expf(v1 - nmax);
      rsum[r] = rsum[r] * corr + half16_sum(e0 + e1);
      o0[r] *= corr;
      o1[r] *= corr;
      rmax[r] = nmax;
      p0[r] = e0; p1[r] = e1;
    }

    // Re-layout P (C layout) -> A layout through LDS
#pragma unroll
    for (int r = 0; r < 8; r++) {
      int row = r + half * 8;
      pbuf[wv][row][li]      = (bf16)p0[r];
      pbuf[wv][row][16 + li] = (bf16)p1[r];
    }
    __syncthreads();
    v16bf ap = loadA16(&pbuf[wv][li][0], 0, half);
    __syncthreads();

    // O += P @ V  (two 16-dim column halves).  vT gives contiguous keys.
    const bf16* vb = vT + ((size_t)(b * DIN_) + h * H_ + li) * N_ + ks + half * 16;
    v16bf bv0 = *(const v16bf*)vb;
    v16bf bv1 = *(const v16bf*)(vb + 16 * N_);
    if (ks + 32 < N_) __builtin_prefetch(vb + 32, 0, 0);
    o0 = wmma_bf16(ap, bv0, o0);
    o1 = wmma_bf16(ap, bv1, o1);
  }

#pragma unroll
  for (int r = 0; r < 8; r++) {
    int has = half16_or(rhas[r]);
    float g = gate ? (has ? 1.f : 0.f) : 1.f;
    float s = g / rsum[r];
    int qrow = qbase + r + half * 8;
    bf16* op = att + ((size_t)(b * N_ + qrow)) * DIN_ + h * H_;
    op[li]      = (bf16)(o0[r] * s);
    op[16 + li] = (bf16)(o1[r] * s);
  }
}

// ------------------------------------------------------------- final kernel
// out = concat(relu(attI@woI+boI), relu(attX@woX+boX)) @ wf + bf   (f32 out)
__global__ void __launch_bounds__(128) final_kernel(
    const bf16* __restrict__ attI, const bf16* __restrict__ attX,
    const bf16* __restrict__ woTI, const float* __restrict__ boI,
    const bf16* __restrict__ woTX, const float* __restrict__ boX,
    const bf16* __restrict__ wfT,  const float* __restrict__ bfinal,
    float* __restrict__ outp) {
  __shared__ __align__(32) bf16 pbuf[4][16][32];

  int lane = threadIdx.x & 31, wv = threadIdx.x >> 5;
  int half = lane >> 4, li = lane & 15;
  int t = blockIdx.x * 4 + wv;
  int b = t >> 5, qbase = (t & 31) * 16;

  v16bf a0, a1;
#pragma unroll
  for (int part = 0; part < 2; part++) {
    const bf16* att = part ? attX : attI;
    const bf16* woT = part ? woTX : woTI;
    const float* bo = part ? boX : boI;

    const bf16* ar = att + ((size_t)(b * N_ + qbase + li)) * DIN_;
    v16bf a[4];
#pragma unroll
    for (int ks = 0; ks < 4; ks++) a[ks] = loadA16(ar, ks * 32, half);

#pragma unroll
    for (int nt = 0; nt < 2; nt++) {
      v8f acc = {};
      int n = nt * 16 + li;  // 0..31
      const bf16* wp = woT + (size_t)n * DIN_ + half * 16;
#pragma unroll
      for (int ks = 0; ks < 4; ks++) {
        v16bf bw = *(const v16bf*)(wp + ks * 32);
        acc = wmma_bf16(a[ks], bw, acc);
      }
      float bb = bo[n];
#pragma unroll
      for (int r = 0; r < 8; r++) {
        float v = acc[r] + bb;
        v = v > 0.f ? v : 0.f;
        pbuf[wv][r + half * 8][n] = (bf16)v;
      }
    }
    __syncthreads();
    v16bf av = loadA16(&pbuf[wv][li][0], 0, half);
    __syncthreads();
    if (part == 0) a0 = av; else a1 = av;
  }

  // out tile: K=64 (inner dims 0-31, inter dims 32-63), N=128
#pragma unroll
  for (int nt = 0; nt < 8; nt++) {
    v8f acc = {};
    int n = nt * 16 + li;
    const bf16* wp = wfT + (size_t)n * 64 + half * 16;
    v16bf b0 = *(const v16bf*)(wp);
    v16bf b1 = *(const v16bf*)(wp + 32);
    acc = wmma_bf16(a0, b0, acc);
    acc = wmma_bf16(a1, b1, acc);
    float bb = bfinal[n];
#pragma unroll
    for (int r = 0; r < 8; r++) {
      int row = qbase + r + half * 8;
      outp[((size_t)(b * N_) + row) * DIN_ + n] = acc[r] + bb;
    }
  }
}

// ----------------------------------------------------------------- launcher
extern "C" void kernel_launch(void* const* d_in, const int* in_sizes, int n_in,
                              void* d_out, int out_size, void* d_ws, size_t ws_size,
                              hipStream_t stream) {
  (void)in_sizes; (void)n_in; (void)out_size; (void)ws_size;

  const float* x0 = (const float*)d_in[0];
  const float* x1 = (const float*)d_in[1];
  const int* m00 = (const int*)d_in[2];
  const int* m01 = (const int*)d_in[3];
  const int* m10 = (const int*)d_in[4];
  const int* m11 = (const int*)d_in[5];
  const float* bq0 = (const float*)d_in[7],  *bk0 = (const float*)d_in[9];
  const float* bv0 = (const float*)d_in[11];
  const float* bq1 = (const float*)d_in[13], *bk1 = (const float*)d_in[15];
  const float* bv1 = (const float*)d_in[17];
  const float* bo00 = (const float*)d_in[19], *bo01 = (const float*)d_in[21];
  const float* bo10 = (const float*)d_in[23], *bo11 = (const float*)d_in[25];
  const float* bf0 = (const float*)d_in[27], *bf1 = (const float*)d_in[29];

  char* ws = (char*)d_ws;
  const size_t SZ = (size_t)B_ * N_ * DIN_ * sizeof(bf16);  // 4 MiB
  bf16* q0   = (bf16*)(ws + 0 * SZ);
  bf16* k0   = (bf16*)(ws + 1 * SZ);
  bf16* vT0  = (bf16*)(ws + 2 * SZ);
  bf16* q1   = (bf16*)(ws + 3 * SZ);
  bf16* k1   = (bf16*)(ws + 4 * SZ);
  bf16* vT1  = (bf16*)(ws + 5 * SZ);
  bf16* a00  = (bf16*)(ws + 6 * SZ);
  bf16* a11  = (bf16*)(ws + 7 * SZ);
  bf16* a01  = (bf16*)(ws + 8 * SZ);
  bf16* a10  = (bf16*)(ws + 9 * SZ);
  char* wp = ws + 10 * SZ;
  bf16* wqT0  = (bf16*)(wp + 0);
  bf16* wkT0  = (bf16*)(wp + 32768);
  bf16* wvT0  = (bf16*)(wp + 65536);
  bf16* wqT1  = (bf16*)(wp + 98304);
  bf16* wkT1  = (bf16*)(wp + 131072);
  bf16* wvT1  = (bf16*)(wp + 163840);
  bf16* woT00 = (bf16*)(wp + 196608);
  bf16* woT01 = (bf16*)(wp + 204800);
  bf16* woT10 = (bf16*)(wp + 212992);
  bf16* woT11 = (bf16*)(wp + 221184);
  bf16* wfT0  = (bf16*)(wp + 229376);
  bf16* wfT1  = (bf16*)(wp + 245760);

  // Weight transposes (f32 -> bf16, [K,N] -> [N,K])
  tr_kernel<<<64, 256, 0, stream>>>((const float*)d_in[6],  wqT0, 128, 128);
  tr_kernel<<<64, 256, 0, stream>>>((const float*)d_in[8],  wkT0, 128, 128);
  tr_kernel<<<64, 256, 0, stream>>>((const float*)d_in[10], wvT0, 128, 128);
  tr_kernel<<<64, 256, 0, stream>>>((const float*)d_in[12], wqT1, 128, 128);
  tr_kernel<<<64, 256, 0, stream>>>((const float*)d_in[14], wkT1, 128, 128);
  tr_kernel<<<64, 256, 0, stream>>>((const float*)d_in[16], wvT1, 128, 128);
  tr_kernel<<<16, 256, 0, stream>>>((const float*)d_in[18], woT00, 128, 32);
  tr_kernel<<<16, 256, 0, stream>>>((const float*)d_in[20], woT01, 128, 32);
  tr_kernel<<<16, 256, 0, stream>>>((const float*)d_in[22], woT10, 128, 32);
  tr_kernel<<<16, 256, 0, stream>>>((const float*)d_in[24], woT11, 128, 32);
  tr_kernel<<<32, 256, 0, stream>>>((const float*)d_in[26], wfT0, 64, 128);
  tr_kernel<<<32, 256, 0, stream>>>((const float*)d_in[28], wfT1, 64, 128);

  // QKV projections (1024 wave-tiles -> 256 blocks of 4 waves)
  qkv_kernel<<<256, 128, 0, stream>>>(x0, wqT0, bq0, q0, 0);
  qkv_kernel<<<256, 128, 0, stream>>>(x0, wkT0, bk0, k0, 0);
  qkv_kernel<<<256, 128, 0, stream>>>(x0, wvT0, bv0, vT0, 1);
  qkv_kernel<<<256, 128, 0, stream>>>(x1, wqT1, bq1, q1, 0);
  qkv_kernel<<<256, 128, 0, stream>>>(x1, wkT1, bk1, k1, 0);
  qkv_kernel<<<256, 128, 0, stream>>>(x1, wvT1, bv1, vT1, 1);

  // Four masked attentions (4096 wave-tiles -> 1024 blocks)
  attn_kernel<<<1024, 128, 0, stream>>>(q0, k0, vT0, m00, a00, 0);
  attn_kernel<<<1024, 128, 0, stream>>>(q1, k1, vT1, m11, a11, 0);
  attn_kernel<<<1024, 128, 0, stream>>>(q0, k1, vT1, m01, a01, 1);
  attn_kernel<<<1024, 128, 0, stream>>>(q1, k0, vT0, m10, a10, 1);

  // Fused wo+relu+concat+wf
  float* out = (float*)d_out;
  final_kernel<<<256, 128, 0, stream>>>(a00, a01, woT00, bo00, woT01, bo01,
                                        wfT0, bf0, out);
  final_kernel<<<256, 128, 0, stream>>>(a11, a10, woT11, bo11, woT10, bo10,
                                        wfT1, bf1, out + (size_t)B_ * N_ * DIN_);
}
